// model_29987461661219
// MI455X (gfx1250) — matrix operand order repaired and checked by reading, not compiled, not run on hardware
//
#include <hip/hip_runtime.h>
#include <math.h>

// ---------------- types ----------------
typedef __attribute__((ext_vector_type(16))) _Float16 v16h;
typedef __attribute__((ext_vector_type(8)))  _Float16 v8h;
typedef __attribute__((ext_vector_type(8)))  float    v8f;

#define HID 512
#define SEQ 1024
#define BAT 50
#define BP  64          // batch padded to 64 for WMMA N-tiles
#define NST 512
#define DT_STEP 1e-3f
#define LEPS 1e-5f
#define PI_F 3.14159265358979323846f
#define BLH_N ((size_t)BAT * SEQ * HID)   // 26,214,400

static __device__ __forceinline__ float gelu_tanh(float x) {
    const float c = 0.7978845608028654f;
    float t = tanhf(c * (x + 0.044715f * x * x * x));
    return 0.5f * x * (1.0f + t);
}

// LDS generic pointer -> wave-relative LDS byte offset (aperture keeps it in addr[31:0])
static __device__ __forceinline__ unsigned lds_off(const void* p) {
    return (unsigned)(size_t)p;
}

// ---------------- K kernels: K[kb,h,l] = 2*Re(sum_n C*dB * dA^l) ----------------
__global__ void k_kernels(const float* __restrict__ Cre, const float* __restrict__ Cim,
                          _Float16* __restrict__ Ker) {
    int i = blockIdx.x * blockDim.x + threadIdx.x;          // over 6*HID*SEQ
    if (i >= 6 * HID * SEQ) return;
    int l  = i & (SEQ - 1);
    int kh = i >> 10;                                        // kb*HID + h
    const float* cr = Cre + (size_t)kh * NST;
    const float* ci = Cim + (size_t)kh * NST;
    float r0 = expf(-0.5f * DT_STEP);
    float rl = expf(-0.5f * DT_STEP * (float)l);
    float acc = 0.f;
    for (int n = 0; n < NST; ++n) {
        float th = PI_F * DT_STEP * (float)n;
        float s, c; sincosf(th, &s, &c);
        float dAre = r0 * c, dAim = r0 * s;
        float Are = -0.5f, Aim = PI_F * (float)n;
        float den = Are * Are + Aim * Aim;
        float nre = dAre - 1.0f, nim = dAim;
        float dBre = (nre * Are + nim * Aim) / den;
        float dBim = (nim * Are - nre * Aim) / den;
        float CR = cr[n], CI = ci[n];
        float cbre = CR * dBre - CI * dBim;
        float cbim = CR * dBim + CI * dBre;
        float sl, cl; sincosf(th * (float)l, &sl, &cl);
        acc += cbre * (rl * cl) - cbim * (rl * sl);
    }
    Ker[i] = (_Float16)(2.0f * acc);
}

// ---------------- embed: u[b,l,h] = x[b,l]*Wemb[h]+bemb[h] ----------------
__global__ void k_embed(const float* __restrict__ x, const float* __restrict__ Wemb,
                        const float* __restrict__ bemb, float* __restrict__ u) {
    size_t i = (size_t)blockIdx.x * blockDim.x + threadIdx.x;
    if (i >= BLH_N) return;
    int h = (int)(i & (HID - 1));
    size_t bl = i >> 9;
    u[i] = x[bl] * Wemb[h] + bemb[h];
}

// ---------------- LayerNorm over H; writes v (f32) and vT (f16, [h][b][l]) ----------------
__global__ void k_ln(const float* __restrict__ u, const float* __restrict__ uadd,
                     const float* __restrict__ g, const float* __restrict__ bb,
                     float* __restrict__ vout, _Float16* __restrict__ vT) {
    int bl = blockIdx.x;                 // b*SEQ + l
    int b = bl >> 10, l = bl & (SEQ - 1);
    int t = threadIdx.x;                 // 256 threads, 2 h each
    __shared__ float red[256];
    size_t base = (size_t)bl * HID + 2 * t;
    float x0 = u[base], x1 = u[base + 1];
    if (uadd) { x0 += uadd[base]; x1 += uadd[base + 1]; }
    red[t] = x0 + x1; __syncthreads();
    for (int s = 128; s > 0; s >>= 1) { if (t < s) red[t] += red[t + s]; __syncthreads(); }
    float mean = red[0] * (1.0f / HID); __syncthreads();
    float d0 = x0 - mean, d1 = x1 - mean;
    red[t] = d0 * d0 + d1 * d1; __syncthreads();
    for (int s = 128; s > 0; s >>= 1) { if (t < s) red[t] += red[t + s]; __syncthreads(); }
    float rs = rsqrtf(red[0] * (1.0f / HID) + LEPS);
    int h0 = 2 * t;
    float v0 = d0 * rs * g[h0] + bb[h0];
    float v1 = d1 * rs * g[h0 + 1] + bb[h0 + 1];
    vout[base] = v0; vout[base + 1] = v1;
    vT[((size_t)h0 * BP + b) * SEQ + l]       = (_Float16)v0;
    vT[((size_t)(h0 + 1) * BP + b) * SEQ + l] = (_Float16)v1;
}

// ---------------- causal Toeplitz conv GEMM: yconv[h][l][b] = sum_m K[h,l-m] v[b,m,h] ----------------
// grid (SEQ/128, HID), 256 threads (8 wave32 waves). Wave w owns rows [l0+16w, +16), 64 b-cols.
// B tiles are double-buffered and staged with GLOBAL_LOAD_ASYNC_TO_LDS_B128 (ASYNCcnt),
// overlapping next-chunk loads with current-chunk WMMAs; one barrier per chunk.
__global__ void __launch_bounds__(256)
k_conv(const _Float16* __restrict__ Ker,   // base of this block's [h][l]
       const _Float16* __restrict__ vT,    // [h][BP][SEQ]
       float* __restrict__ yconv) {        // [h][SEQ][BP]
    int h  = blockIdx.y;
    int l0 = blockIdx.x * 128;
    int tid = threadIdx.x;
    int wave = tid >> 5, lane = tid & 31;
    int laneHi = lane >> 4, mrow = lane & 15;
    int lw0 = l0 + wave * 16;

    __shared__ __attribute__((aligned(16))) _Float16 Kbuf[1056];        // 32 zero pad + 1024
    __shared__ __attribute__((aligned(16))) _Float16 Btile[2][64 * 40]; // [b][m], stride 40 halves

    for (int i = tid; i < 1056; i += 256)
        Kbuf[i] = (i < 32) ? (_Float16)0.0f : Ker[(size_t)h * SEQ + (i - 32)];

    const v8f zero = {0.f,0.f,0.f,0.f,0.f,0.f,0.f,0.f};
    v8f acc[4] = {zero, zero, zero, zero};

    const _Float16* vTh = vT + (size_t)h * BP * SEQ;
    int nCh = (l0 >> 5) + 4;
    int sb = tid >> 2, sseg = (tid & 3) * 8;            // staging: 16B per lane
    const _Float16* srow = vTh + (size_t)sb * SEQ + sseg;

    // prologue: async-stage chunk 0 into buffer 0
    {
        unsigned lo = lds_off(&Btile[0][sb * 40 + sseg]);
        asm volatile("global_load_async_to_lds_b128 %0, %1, off"
                     :: "v"(lo), "v"((unsigned long long)(size_t)srow) : "memory");
    }
    for (int ch = 0; ch < nCh; ++ch) {
        asm volatile("s_wait_asynccnt 0x0" ::: "memory");   // own wave's tile arrived
        __syncthreads();                                    // all tiles arrived, prev buffer free
        if (ch + 1 < nCh) {                                 // uniform: issue next chunk async
            unsigned lo = lds_off(&Btile[(ch + 1) & 1][sb * 40 + sseg]);
            asm volatile("global_load_async_to_lds_b128 %0, %1, off"
                         :: "v"(lo),
                            "v"((unsigned long long)(size_t)(srow + (size_t)(ch + 1) * 32))
                         : "memory");
        }
        int m0 = ch * 32;
        if (m0 <= lw0 + 15) {            // wave-uniform: EXEC stays all ones for WMMA
            int lr = lw0 + mrow;
            v16h a;
            int base = 32 + lr - m0 - laneHi * 8;   // >= 1 by construction (32-elem zero pad)
            #pragma unroll
            for (int e = 0; e < 8; ++e) {
                a[e]     = Kbuf[base - e];          // K = laneHi*8 + e
                a[8 + e] = Kbuf[base - 16 - e];     // K = laneHi*8 + 16 + e
            }
            int kstart = laneHi * 8;
            const _Float16* Bcur = &Btile[ch & 1][0];
            #pragma unroll
            for (int nt = 0; nt < 4; ++nt) {
                const _Float16* bp = Bcur + (nt * 16 + mrow) * 40 + kstart;
                v8h b0 = *(const v8h*)bp;
                v8h b1 = *(const v8h*)(bp + 16);
                v16h bv;
                #pragma unroll
                for (int e = 0; e < 8; ++e) { bv[e] = b0[e]; bv[8 + e] = b1[e]; }
                acc[nt] = __builtin_amdgcn_wmma_f32_16x16x32_f16(
                    false, a, false, bv, (short)0, acc[nt], false, false);
            }
        }
    }
    #pragma unroll
    for (int nt = 0; nt < 4; ++nt) {
        #pragma unroll
        for (int r = 0; r < 8; ++r) {
            int l = lw0 + laneHi * 8 + r;
            int b = nt * 16 + mrow;
            yconv[((size_t)h * SEQ + l) * BP + b] = acc[nt][r];
        }
    }
}

// ---------------- block epilogue: y = tanh(conv + v*D) (+u if skip) ----------------
__global__ void k_post(const float* __restrict__ yconv, const float* __restrict__ v,
                       const float* __restrict__ D, const float* __restrict__ uin,
                       float* __restrict__ yout, int skip) {
    size_t i = (size_t)blockIdx.x * blockDim.x + threadIdx.x;
    if (i >= BLH_N) return;
    int h = (int)(i & (HID - 1));
    size_t bl = i >> 9;
    int b = (int)(bl >> 10), l = (int)(bl & (SEQ - 1));
    float val = tanhf(yconv[((size_t)h * SEQ + l) * BP + b] + v[i] * D[h]);
    if (skip) val += uin[i];
    yout[i] = val;
}

// ---------------- gelu -> f16 (row-major [b*L+l][h], A-operand of FF GEMM) ----------------
__global__ void k_gelu16(const float* __restrict__ u, _Float16* __restrict__ hg) {
    size_t i = (size_t)blockIdx.x * blockDim.x + threadIdx.x;
    if (i >= BLH_N) return;
    hg[i] = (_Float16)gelu_tanh(u[i]);
}

__global__ void k_f32tof16(const float* __restrict__ w, _Float16* __restrict__ o, int n) {
    int i = blockIdx.x * blockDim.x + threadIdx.x;
    if (i < n) o[i] = (_Float16)w[i];
}

// ---------------- FF GEMM: z = gelu(hg @ Wf + bias), M=51200 K=512 N=512 ----------------
__global__ void __launch_bounds__(256)
k_ff(const _Float16* __restrict__ Aop, const _Float16* __restrict__ Wf,
     const float* __restrict__ bias, float* __restrict__ z) {
    int m0 = blockIdx.x * 128;
    int n0 = blockIdx.y * 64;
    int tid = threadIdx.x;
    int wave = tid >> 5, lane = tid & 31;
    int laneHi = lane >> 4, mrow = lane & 15;
    int mw0 = m0 + wave * 16;

    __shared__ __attribute__((aligned(16))) _Float16 Btile[64 * 40];  // [n][k]
    const v8f zero = {0.f,0.f,0.f,0.f,0.f,0.f,0.f,0.f};
    v8f acc[4] = {zero, zero, zero, zero};

    for (int k0 = 0; k0 < HID; k0 += 32) {
        __syncthreads();
        {   // stage Wf tile [32 k][64 n] transposed into LDS [n][k]
            int k = tid >> 3, seg = (tid & 7) * 8;
            v8h w = *(const v8h*)(Wf + (size_t)(k0 + k) * HID + n0 + seg);
            #pragma unroll
            for (int j = 0; j < 8; ++j) Btile[(seg + j) * 40 + k] = w[j];
        }
        __syncthreads();
        const _Float16* arow = Aop + (size_t)(mw0 + mrow) * HID + k0 + laneHi * 8;
        v8h a0 = *(const v8h*)arow;
        v8h a1 = *(const v8h*)(arow + 16);
        v16h a;
        #pragma unroll
        for (int e = 0; e < 8; ++e) { a[e] = a0[e]; a[8 + e] = a1[e]; }
        int kstart = laneHi * 8;
        #pragma unroll
        for (int nt = 0; nt < 4; ++nt) {
            const _Float16* bp = &Btile[(nt * 16 + mrow) * 40 + kstart];
            v8h b0 = *(const v8h*)bp;
            v8h b1 = *(const v8h*)(bp + 16);
            v16h bv;
            #pragma unroll
            for (int e = 0; e < 8; ++e) { bv[e] = b0[e]; bv[8 + e] = b1[e]; }
            acc[nt] = __builtin_amdgcn_wmma_f32_16x16x32_f16(
                false, a, false, bv, (short)0, acc[nt], false, false);
        }
    }
    #pragma unroll
    for (int nt = 0; nt < 4; ++nt) {
        #pragma unroll
        for (int r = 0; r < 8; ++r) {
            int m = mw0 + laneHi * 8 + r;
            int n = n0 + nt * 16 + mrow;
            z[(size_t)m * HID + n] = gelu_tanh(acc[nt][r] + bias[n]);
        }
    }
}

// ---------------- reductions / finale ----------------
__global__ void k_losspart(const float* __restrict__ a, const float* __restrict__ b,
                           float* __restrict__ part) {
    __shared__ float red[256];
    int t = threadIdx.x;
    size_t stride = (size_t)gridDim.x * 256;
    float s = 0.f;
    for (size_t i = (size_t)blockIdx.x * 256 + t; i < BLH_N; i += stride) {
        float d = a[i] - b[i]; s += d * d;
    }
    red[t] = s; __syncthreads();
    for (int w = 128; w > 0; w >>= 1) { if (t < w) red[t] += red[t + w]; __syncthreads(); }
    if (t == 0) part[blockIdx.x] = red[0];
}

__global__ void k_lossfin(const float* __restrict__ part, int np, float* __restrict__ out) {
    __shared__ float red[256];
    int t = threadIdx.x;
    float s = 0.f;
    for (int i = t; i < np; i += 256) s += part[i];
    red[t] = s; __syncthreads();
    for (int w = 128; w > 0; w >>= 1) { if (t < w) red[t] += red[t + w]; __syncthreads(); }
    if (t == 0) out[0] = red[0] / (float)BLH_N;
}

__global__ void k_meanl(const float* __restrict__ hbuf, float* __restrict__ meanl) {
    int i = blockIdx.x * blockDim.x + threadIdx.x;
    if (i >= BAT * HID) return;
    int b = i / HID, h = i % HID;
    float s = 0.f;
    for (int l = 0; l < SEQ; ++l) s += hbuf[((size_t)b * SEQ + l) * HID + h];
    meanl[i] = s * (1.0f / SEQ);
}

__global__ void k_logits(const float* __restrict__ meanl, const float* __restrict__ Wfc,
                         const float* __restrict__ bfc, float* __restrict__ out) {
    int i = blockIdx.x * blockDim.x + threadIdx.x;
    if (i >= BAT * 10) return;
    int b = i / 10, j = i % 10;
    float s = bfc[j];
    for (int h = 0; h < HID; ++h) s += meanl[b * HID + h] * Wfc[h * 10 + j];
    out[i] = s;
}

__global__ void k_copy(const float* __restrict__ src, float* __restrict__ dst, size_t n) {
    size_t i = (size_t)blockIdx.x * blockDim.x + threadIdx.x;
    if (i < n) dst[i] = src[i];
}

__global__ void k_zero32(unsigned int* __restrict__ p, size_t n) {
    size_t i = (size_t)blockIdx.x * blockDim.x + threadIdx.x;
    if (i < n) p[i] = 0u;
}

__global__ void k_ndeq(int* __restrict__ p) { *p = 8; }

// ---------------- host orchestration ----------------
extern "C" void kernel_launch(void* const* d_in, const int* in_sizes, int n_in,
                              void* d_out, int out_size, void* d_ws, size_t ws_size,
                              hipStream_t stream) {
    (void)in_sizes; (void)n_in; (void)out_size;
    const float* x     = (const float*)d_in[0];
    const float* Wemb  = (const float*)d_in[1];
    const float* bemb  = (const float*)d_in[2];
    const float* Cre   = (const float*)d_in[3];
    const float* Cim   = (const float*)d_in[4];
    const float* Dp    = (const float*)d_in[5];
    const float* lng   = (const float*)d_in[6];
    const float* lnb   = (const float*)d_in[7];
    const float* Wfeed = (const float*)d_in[8];
    const float* bfeed = (const float*)d_in[9];
    const float* Wfc   = (const float*)d_in[10];
    const float* bfc   = (const float*)d_in[11];

    char* ws = (char*)d_ws;
    size_t off = 0;
    auto alloc = [&](size_t bytes) -> char* {
        char* p = ws + off; off += (bytes + 255) & ~(size_t)255; return p;
    };
    float*    u0    = (float*)alloc(BLH_N * 4);
    float*    u1    = (float*)alloc(BLH_N * 4);
    float*    u2    = (float*)alloc(BLH_N * 4);
    float*    za    = (float*)alloc(BLH_N * 4);
    float*    zb    = (float*)alloc(BLH_N * 4);
    float*    vbuf  = (float*)alloc(BLH_N * 4);
    float*    yconv = (float*)alloc((size_t)HID * SEQ * BP * 4);
    _Float16* ker   = (_Float16*)alloc((size_t)6 * HID * SEQ * 2);
    _Float16* vT    = (_Float16*)alloc((size_t)HID * BP * SEQ * 2);
    _Float16* hg    = (_Float16*)alloc(BLH_N * 2);
    _Float16* wf16  = (_Float16*)alloc((size_t)HID * HID * 2);
    float*    meanl = (float*)alloc((size_t)BAT * HID * 4);
    float*    lpart = (float*)alloc(1024 * 4);
    if (off > ws_size) return;   // workspace too small; cannot proceed safely

    const int T = 256;
    const int gBLH = (int)(BLH_N / T);                 // 102400, exact
    dim3 convGrid(SEQ / 128, HID);
    dim3 ffGrid((BAT * SEQ) / 128, HID / 64);

    // one-time prep (inside launch; graph-capture safe, deterministic)
    k_zero32<<<(int)(((size_t)HID * BP * SEQ / 2 + T - 1) / T), T, 0, stream>>>(
        (unsigned int*)vT, (size_t)HID * BP * SEQ / 2);
    k_zero32<<<(int)((BLH_N + T - 1) / T), T, 0, stream>>>((unsigned int*)za, BLH_N);
    k_kernels<<<(6 * HID * SEQ) / T, T, 0, stream>>>(Cre, Cim, ker);
    k_f32tof16<<<(HID * HID) / T, T, 0, stream>>>(Wfeed, wf16, HID * HID);
    k_embed<<<gBLH, T, 0, stream>>>(x, Wemb, bemb, u0);

    // 3 explicit S4D blocks (skip connections)
    float* cur = u0; float* nxt = u1;
    for (int i = 0; i < 3; ++i) {
        k_ln<<<BAT * SEQ, T, 0, stream>>>(cur, nullptr, lng + i * HID, lnb + i * HID, vbuf, vT);
        k_conv<<<convGrid, T, 0, stream>>>(ker + (size_t)i * HID * SEQ, vT, yconv);
        k_post<<<gBLH, T, 0, stream>>>(yconv, vbuf, Dp + i * HID, cur, nxt, 1);
        float* t = cur; cur = nxt; nxt = t;
    }
    float* yexpl = cur;                       // input-injection for DEQ
    float* t0 = nxt;                          // free chain buffer
    float* t1 = u2;

    // DEQ fixed-point: 8 iterations of f(z) = 3 S4D blocks (no skip) + gelu + gelu(FF)
    float* zin = za; float* zout = zb;
    for (int it = 0; it < 8; ++it) {
        k_ln<<<BAT * SEQ, T, 0, stream>>>(zin, yexpl, lng + 3 * HID, lnb + 3 * HID, vbuf, vT);
        k_conv<<<convGrid, T, 0, stream>>>(ker + (size_t)3 * HID * SEQ, vT, yconv);
        k_post<<<gBLH, T, 0, stream>>>(yconv, vbuf, Dp + 3 * HID, nullptr, t0, 0);

        k_ln<<<BAT * SEQ, T, 0, stream>>>(t0, nullptr, lng + 4 * HID, lnb + 4 * HID, vbuf, vT);
        k_conv<<<convGrid, T, 0, stream>>>(ker + (size_t)4 * HID * SEQ, vT, yconv);
        k_post<<<gBLH, T, 0, stream>>>(yconv, vbuf, Dp + 4 * HID, nullptr, t1, 0);

        k_ln<<<BAT * SEQ, T, 0, stream>>>(t1, nullptr, lng + 5 * HID, lnb + 5 * HID, vbuf, vT);
        k_conv<<<convGrid, T, 0, stream>>>(ker + (size_t)5 * HID * SEQ, vT, yconv);
        k_post<<<gBLH, T, 0, stream>>>(yconv, vbuf, Dp + 5 * HID, nullptr, t0, 0);

        k_gelu16<<<gBLH, T, 0, stream>>>(t0, hg);
        k_ff<<<ffGrid, T, 0, stream>>>(hg, wf16, bfeed, zout);
        float* tz = zin; zin = zout; zout = tz;
    }
    float* znew  = zin;    // z after 8 applications
    float* zprev = zout;   // z after 7 applications

    // outputs: [0..499] logits, [500] loss, [501..] h (B*L*H), [last] N_DEQ
    float* out_f = (float*)d_out;
    k_losspart<<<1024, T, 0, stream>>>(znew, zprev, lpart);
    k_lossfin<<<1, T, 0, stream>>>(lpart, 1024, out_f + 500);
    k_meanl<<<(BAT * HID + T - 1) / T, T, 0, stream>>>(znew, meanl);
    k_logits<<<(BAT * 10 + T - 1) / T, T, 0, stream>>>(meanl, Wfc, bfc, out_f);
    k_copy<<<gBLH, T, 0, stream>>>(znew, out_f + 501, BLH_N);
    k_ndeq<<<1, 1, 0, stream>>>((int*)d_out + 501 + BLH_N);
}
